// SelfAttention_20555713479060
// MI455X (gfx1250) — matrix operand order repaired and checked by reading, not compiled, not run on hardware
//
#include <hip/hip_runtime.h>
#include <math.h>

// ---------------------------------------------------------------------------
// MI455X (gfx1250) self-attention forward:
//   qkv = x @ W_qkv ; rmsnorm(q,k) ; flash-attention ; out = attn @ W_proj + b
// Matmuls via v_wmma_f32_16x16x32_bf16 (wave32, fp32 accumulate).
// K/Q tile staging in the attention kernel uses the Tensor Data Mover
// (tensor_load_to_lds + s_wait_tensorcnt) when the builtin is available.
// ---------------------------------------------------------------------------

typedef __attribute__((ext_vector_type(16))) __bf16 v16bf;
typedef __attribute__((ext_vector_type(8)))  __bf16 v8bf;
typedef __attribute__((ext_vector_type(4)))  __bf16 v4bf;
typedef __attribute__((ext_vector_type(2)))  __bf16 v2bf;
typedef __attribute__((ext_vector_type(8)))  float  v8f;

typedef unsigned int u32x4 __attribute__((ext_vector_type(4)));
typedef int          i32x4 __attribute__((ext_vector_type(4)));
typedef int          i32x8 __attribute__((ext_vector_type(8)));

#define BATCH   2
#define HEADS   16
#define LSEQ    2048
#define DMODEL  1024
#define DHEAD   64

#if __has_builtin(__builtin_amdgcn_tensor_load_to_lds)
#define HAVE_TDM 1
#else
#define HAVE_TDM 0
#endif

static __device__ __forceinline__ v8f zero_v8f() {
  v8f z;
#pragma unroll
  for (int i = 0; i < 8; ++i) z[i] = 0.0f;
  return z;
}

static __device__ __forceinline__ v16bf frag_cat(v8bf lo, v8bf hi) {
  return __builtin_shufflevector(lo, hi, 0,1,2,3,4,5,6,7,8,9,10,11,12,13,14,15);
}

static __device__ __forceinline__ v8f wmma_bf16(v16bf a, v16bf b, v8f c) {
  // (neg_a, A, neg_b, B, c_mod, C, reuse_a, reuse_b)
  return __builtin_amdgcn_wmma_f32_16x16x32_bf16(false, a, false, b, (short)0, c,
                                                 false, false);
}

static __device__ __forceinline__ void wait_tensorcnt0() {
#if __has_builtin(__builtin_amdgcn_s_wait_tensorcnt)
  __builtin_amdgcn_s_wait_tensorcnt(0);
#else
  asm volatile("s_wait_tensorcnt 0x0" ::: "memory");
#endif
}

// Issue a TDM 2-D tile load (bf16 elements): tile_dim0 x tile_dim1 elements,
// row stride `stride0` elements, packed contiguously at LDS byte addr lds_off.
static __device__ __forceinline__ void tdm_load_tile_2d(const void* gsrc,
                                                        unsigned lds_off,
                                                        unsigned dim0, unsigned dim1,
                                                        unsigned stride0,
                                                        unsigned tile0, unsigned tile1) {
#if HAVE_TDM
  unsigned long long ga = (unsigned long long)gsrc;
  // ---- D# group 0: count=1 (valid user descriptor), lds_addr, global_addr,
  //      type=2 ("image") in bits[127:126]
  u32x4 g0;
  g0[0] = 1u;
  g0[1] = lds_off;
  g0[2] = (unsigned)(ga & 0xffffffffu);
  g0[3] = (unsigned)((ga >> 32) & 0x01ffffffu) | (2u << 30);
  // ---- D# group 1: wg_mask=0 (not in a cluster), data_size=1 (2 bytes),
  //      tensor_dim0/1, tile_dim0/1, tensor_dim0_stride
  i32x8 g1;
  g1[0] = (int)(1u << 16);                                   // data_size = 2B
  g1[1] = (int)((dim0 & 0xffffu) << 16);                     // tensor_dim0 lo
  g1[2] = (int)(((dim0 >> 16) & 0xffffu) | ((dim1 & 0xffffu) << 16));
  g1[3] = (int)(((dim1 >> 16) & 0xffffu) | ((tile0 & 0xffffu) << 16));
  g1[4] = (int)(tile1 & 0xffffu);                            // tile_dim1, tile_dim2=0
  g1[5] = (int)stride0;                                      // dim0 stride lo32
  g1[6] = 0;
  g1[7] = 0;
  i32x4 gz = {0, 0, 0, 0};                                   // groups 2/3: 2-D tensor
#if __clang_major__ >= 23
  i32x8 z8 = {0, 0, 0, 0, 0, 0, 0, 0};
  __builtin_amdgcn_tensor_load_to_lds(g0, g1, gz, gz, z8, 0);
#else
  __builtin_amdgcn_tensor_load_to_lds(g0, g1, gz, gz, 0);
#endif
#endif
}

// ---------------------------------------------------------------------------
// fp32 -> bf16 conversion (vectorized x4)
// ---------------------------------------------------------------------------
__global__ __launch_bounds__(256)
void cvt_f32_bf16(const float* __restrict__ src, __bf16* __restrict__ dst, int n4) {
  int i = blockIdx.x * blockDim.x + threadIdx.x;
  int stride = gridDim.x * blockDim.x;
  const float4* s4 = (const float4*)src;
  for (; i < n4; i += stride) {
    float4 v = s4[i];
    v4bf o;
    o[0] = (__bf16)v.x; o[1] = (__bf16)v.y; o[2] = (__bf16)v.z; o[3] = (__bf16)v.w;
    *(v4bf*)(dst + 4 * (size_t)i) = o;
  }
}

// ---------------------------------------------------------------------------
// QKV GEMM: [4096 x 1024] @ [1024 x 3072]; block tile 128(M) x 64(N), 4 waves,
// each wave owns a 32x64 output tile (8 WMMA accumulators), K-chunks of 32.
// Epilogue: RMSNorm for q/k columns, store q,k,v as bf16 in [B,H,L,DH].
// ---------------------------------------------------------------------------
__global__ __launch_bounds__(128)
void attn_qkv_gemm(const __bf16* __restrict__ xb, const __bf16* __restrict__ wb,
                   const float* __restrict__ q_scale, const float* __restrict__ k_scale,
                   __bf16* __restrict__ qout, __bf16* __restrict__ kout,
                   __bf16* __restrict__ vout) {
  __shared__ __bf16 As[128 * 32];   // A tile, row-major [m][k]
  __shared__ __bf16 BsT[64 * 32];   // B tile, transposed [n][k]
  __shared__ float  Cs[128 * 64];   // fp32 epilogue tile

  const int tid  = threadIdx.x;
  const int wave = tid >> 5;
  const int lane = tid & 31;
  const int n0   = blockIdx.x * 64;
  const int m0   = blockIdx.y * 128;

  v8f acc[2][4];
#pragma unroll
  for (int mt = 0; mt < 2; ++mt)
#pragma unroll
    for (int nt = 0; nt < 4; ++nt) acc[mt][nt] = zero_v8f();

  const unsigned int* xu  = (const unsigned int*)xb;
  unsigned int*       AsU = (unsigned int*)As;
  const v2bf*         wb2 = (const v2bf*)wb;

  for (int kt = 0; kt < 32; ++kt) {
    __syncthreads();
    // A: 128x32 bf16 = 2048 dwords, coalesced dword copy
#pragma unroll
    for (int j = 0; j < 16; ++j) {
      int u = tid + j * 128;
      int row = u >> 4, c2 = u & 15;
      AsU[u] = xu[(size_t)(m0 + row) * 512 + (size_t)kt * 16 + c2];
    }
    // B: 32x64 bf16 loaded as dwords, stored transposed [n][k]
#pragma unroll
    for (int j = 0; j < 8; ++j) {
      int e = tid + j * 128;                  // 1024 dwords
      int row = e >> 5, c2 = e & 31;          // row 0..31, col pair 0..31
      v2bf w = wb2[(size_t)(kt * 32 + row) * 1536 + (n0 >> 1) + c2];
      BsT[(2 * c2 + 0) * 32 + row] = w[0];
      BsT[(2 * c2 + 1) * 32 + row] = w[1];
    }
    __syncthreads();

    const int kb_ = (lane >> 4) * 16;
    v16bf afrag[2];
#pragma unroll
    for (int mt = 0; mt < 2; ++mt) {
      int m = wave * 32 + mt * 16 + (lane & 15);
      afrag[mt] = frag_cat(*(const v8bf*)&As[m * 32 + kb_],
                           *(const v8bf*)&As[m * 32 + kb_ + 8]);
    }
#pragma unroll
    for (int nt = 0; nt < 4; ++nt) {
      int nl = nt * 16 + (lane & 15);
      v16bf bfrag = *(const v16bf*)&BsT[nl * 32 + (lane >> 4) * 16];
#pragma unroll
      for (int mt = 0; mt < 2; ++mt)
        acc[mt][nt] = wmma_bf16(afrag[mt], bfrag, acc[mt][nt]);
    }
  }

  // spill accumulators to LDS for cross-lane epilogue
#pragma unroll
  for (int mt = 0; mt < 2; ++mt)
#pragma unroll
    for (int nt = 0; nt < 4; ++nt)
#pragma unroll
      for (int r = 0; r < 8; ++r) {
        int ml = wave * 32 + mt * 16 + r + 8 * (lane >> 4);
        int nl = nt * 16 + (lane & 15);
        Cs[ml * 64 + nl] = acc[mt][nt][r];
      }
  __syncthreads();

  const int kcomp = n0 >> 10;           // 0=q, 1=k, 2=v
  const int h     = (n0 & 1023) >> 6;   // head
  if (kcomp == 2) {
    for (int e = tid; e < 128 * 64; e += 128) {
      int rr = e >> 6, j = e & 63;
      int m = m0 + rr, b = m >> 11, l = m & 2047;
      vout[((size_t)(b * HEADS + h) * LSEQ + l) * DHEAD + j] = (__bf16)Cs[e];
    }
  } else {
    const float* scale = (kcomp == 0) ? q_scale : k_scale;
    __bf16* dst        = (kcomp == 0) ? qout : kout;
    // one thread per output row (128 rows, 128 threads)
    int m = m0 + tid, b = m >> 11, l = m & 2047;
    float ss = 0.0f;
#pragma unroll 8
    for (int j = 0; j < 64; ++j) { float c = Cs[tid * 64 + j]; ss += c * c; }
    float rrms = rsqrtf(ss * (1.0f / 64.0f) + 1e-6f);
    size_t base = ((size_t)(b * HEADS + h) * LSEQ + l) * DHEAD;
#pragma unroll 8
    for (int j = 0; j < 64; ++j)
      dst[base + j] = (__bf16)(Cs[tid * 64 + j] * rrms * scale[j]);
  }
}

// ---------------------------------------------------------------------------
// Flash attention: one block per (b,h, 64 q-rows); online softmax over 64-key
// tiles; per wave 16 q-rows; scores + PV via WMMA bf16. Q and K tiles staged
// into LDS by the Tensor Data Mover when available.
// ---------------------------------------------------------------------------
__global__ __launch_bounds__(128)
void attn_flash(const __bf16* __restrict__ qbuf, const __bf16* __restrict__ kbuf,
                const __bf16* __restrict__ vbuf, __bf16* __restrict__ attnb) {
  __shared__ __bf16 Qs[64 * 64];       // [qrow][dh]
  __shared__ __bf16 Ks[64 * 64];       // [key][dh]
  __shared__ __bf16 VsT[64 * 64];      // [dh][key]
  __shared__ __bf16 Ps[4 * 16 * 64];   // per-wave P tile [m][key]

  const int tid  = threadIdx.x;
  const int wave = tid >> 5;
  const int lane = tid & 31;
  const int bh   = blockIdx.y;
  const int b    = bh >> 4;
  const int h    = bh & 15;
  const int q0   = blockIdx.x * 64;

#if HAVE_TDM
  if (wave == 0) {
    tdm_load_tile_2d(qbuf + ((size_t)bh * LSEQ + q0) * DHEAD,
                     (unsigned)(unsigned long long)(const void*)Qs,
                     64, 64, 64, 64, 64);
  }
#else
  {
    const unsigned int* qsrc =
        (const unsigned int*)(qbuf + ((size_t)bh * LSEQ + q0) * DHEAD);
    unsigned int* QsU = (unsigned int*)Qs;
#pragma unroll
    for (int j = 0; j < 16; ++j) QsU[tid + j * 128] = qsrc[tid + j * 128];
  }
#endif

  float rowmax[8], rowsum[8];
  v8f o_acc[4];
#pragma unroll
  for (int r = 0; r < 8; ++r) { rowmax[r] = -1e30f; rowsum[r] = 0.0f; }
#pragma unroll
  for (int t = 0; t < 4; ++t) o_acc[t] = zero_v8f();

  const int qm = wave * 16 + (lane & 15);

  for (int kv0 = 0; kv0 < LSEQ; kv0 += 64) {
    __syncthreads();  // previous K/V tiles fully consumed
#if HAVE_TDM
    if (wave == 0) {
      tdm_load_tile_2d(kbuf + ((size_t)bh * LSEQ + kv0) * DHEAD,
                       (unsigned)(unsigned long long)(const void*)Ks,
                       64, 64, 64, 64, 64);
    }
#else
    {
      const unsigned int* ksrc =
          (const unsigned int*)(kbuf + ((size_t)bh * LSEQ + kv0) * DHEAD);
      unsigned int* KsU = (unsigned int*)Ks;
#pragma unroll
      for (int j = 0; j < 16; ++j) KsU[tid + j * 128] = ksrc[tid + j * 128];
    }
#endif
    {
      // V tile: dword loads, transposed scalar stores [dh][key]
      const v2bf* vsrc = (const v2bf*)(vbuf + ((size_t)bh * LSEQ + kv0) * DHEAD);
#pragma unroll
      for (int j = 0; j < 16; ++j) {
        int e = tid + j * 128;                // 2048 dwords
        int key = e >> 5, d2 = e & 31;
        v2bf w = vsrc[key * 32 + d2];
        VsT[(2 * d2 + 0) * 64 + key] = w[0];
        VsT[(2 * d2 + 1) * 64 + key] = w[1];
      }
    }
#if HAVE_TDM
    if (wave == 0) wait_tensorcnt0();   // TDM done before the block barrier
#endif
    __syncthreads();

    // ---- S = (Q K^T) * 1/sqrt(DH) ----
    v8f s[4];
#pragma unroll
    for (int nt = 0; nt < 4; ++nt) s[nt] = zero_v8f();
#pragma unroll
    for (int c = 0; c < 2; ++c) {
      int kb_ = c * 32 + (lane >> 4) * 16;
      v16bf afrag = frag_cat(*(const v8bf*)&Qs[qm * 64 + kb_],
                             *(const v8bf*)&Qs[qm * 64 + kb_ + 8]);
      int kk = c * 32 + (lane >> 4) * 16;
#pragma unroll
      for (int nt = 0; nt < 4; ++nt) {
        int n = nt * 16 + (lane & 15);
        v16bf bfrag = *(const v16bf*)&Ks[n * 64 + kk];
        s[nt] = wmma_bf16(afrag, bfrag, s[nt]);
      }
    }
#pragma unroll
    for (int nt = 0; nt < 4; ++nt)
#pragma unroll
      for (int r = 0; r < 8; ++r) s[nt][r] *= 0.125f;

    // ---- online softmax (row stats live per-lane for 8 rows) ----
    float nmax[8], alpha[8], psum[8];
#pragma unroll
    for (int r = 0; r < 8; ++r) {
      float mx = fmaxf(fmaxf(s[0][r], s[1][r]), fmaxf(s[2][r], s[3][r]));
      mx = fmaxf(mx, __shfl_xor(mx, 1, 32));
      mx = fmaxf(mx, __shfl_xor(mx, 2, 32));
      mx = fmaxf(mx, __shfl_xor(mx, 4, 32));
      mx = fmaxf(mx, __shfl_xor(mx, 8, 32));
      nmax[r]  = fmaxf(rowmax[r], mx);
      alpha[r] = __expf(rowmax[r] - nmax[r]);
      rowmax[r] = nmax[r];
      psum[r]  = 0.0f;
    }
#pragma unroll
    for (int nt = 0; nt < 4; ++nt)
#pragma unroll
      for (int r = 0; r < 8; ++r) {
        float p = __expf(s[nt][r] - nmax[r]);
        psum[r] += p;
        Ps[wave * 1024 + (r + 8 * (lane >> 4)) * 64 + nt * 16 + (lane & 15)] =
            (__bf16)p;
      }
#pragma unroll
    for (int r = 0; r < 8; ++r) {
      float sm = psum[r];
      sm += __shfl_xor(sm, 1, 32);
      sm += __shfl_xor(sm, 2, 32);
      sm += __shfl_xor(sm, 4, 32);
      sm += __shfl_xor(sm, 8, 32);
      rowsum[r] = rowsum[r] * alpha[r] + sm;
    }
#pragma unroll
    for (int t = 0; t < 4; ++t)
#pragma unroll
      for (int r = 0; r < 8; ++r) o_acc[t][r] *= alpha[r];

    // ---- O += P V ----
#pragma unroll
    for (int c = 0; c < 2; ++c) {
      int kb_ = c * 32 + (lane >> 4) * 16;
      const __bf16* pp = &Ps[wave * 1024 + (lane & 15) * 64 + kb_];
      v16bf pfrag = frag_cat(*(const v8bf*)pp, *(const v8bf*)(pp + 8));
      int kk = c * 32 + (lane >> 4) * 16;
#pragma unroll
      for (int ot = 0; ot < 4; ++ot) {
        int n = ot * 16 + (lane & 15);
        v16bf vfrag = *(const v16bf*)&VsT[n * 64 + kk];
        o_acc[ot] = wmma_bf16(pfrag, vfrag, o_acc[ot]);
      }
    }
  }

  // ---- normalize, store to attn buffer [B*L, D] (col = h*64 + dh) ----
#pragma unroll
  for (int r = 0; r < 8; ++r) rowsum[r] = 1.0f / rowsum[r];
#pragma unroll
  for (int ot = 0; ot < 4; ++ot)
#pragma unroll
    for (int r = 0; r < 8; ++r) {
      int qrow = q0 + wave * 16 + r + 8 * (lane >> 4);
      int dh   = ot * 16 + (lane & 15);
      attnb[((size_t)(b * LSEQ + qrow)) * DMODEL + h * DHEAD + dh] =
          (__bf16)(o_acc[ot][r] * rowsum[r]);
    }
}

// ---------------------------------------------------------------------------
// Output projection: [4096 x 1024] @ [1024 x 1024] + bias, fp32 out.
// Block tile 128(M) x 64(N); per-wave 32x64 (8 accumulators).
// ---------------------------------------------------------------------------
__global__ __launch_bounds__(128)
void attn_proj_gemm(const __bf16* __restrict__ ab, const __bf16* __restrict__ wb,
                    const float* __restrict__ bias, float* __restrict__ out) {
  __shared__ __bf16 As[128 * 32];
  __shared__ __bf16 BsT[64 * 32];

  const int tid  = threadIdx.x;
  const int wave = tid >> 5;
  const int lane = tid & 31;
  const int n0   = blockIdx.x * 64;
  const int m0   = blockIdx.y * 128;

  v8f acc[2][4];
#pragma unroll
  for (int mt = 0; mt < 2; ++mt)
#pragma unroll
    for (int nt = 0; nt < 4; ++nt) acc[mt][nt] = zero_v8f();

  const unsigned int* au  = (const unsigned int*)ab;
  unsigned int*       AsU = (unsigned int*)As;
  const v2bf*         wb2 = (const v2bf*)wb;

  for (int kt = 0; kt < 32; ++kt) {
    __syncthreads();
#pragma unroll
    for (int j = 0; j < 16; ++j) {
      int u = tid + j * 128;
      int row = u >> 4, c2 = u & 15;
      AsU[u] = au[(size_t)(m0 + row) * 512 + (size_t)kt * 16 + c2];
    }
#pragma unroll
    for (int j = 0; j < 8; ++j) {
      int e = tid + j * 128;
      int row = e >> 5, c2 = e & 31;
      v2bf w = wb2[(size_t)(kt * 32 + row) * 512 + (n0 >> 1) + c2];
      BsT[(2 * c2 + 0) * 32 + row] = w[0];
      BsT[(2 * c2 + 1) * 32 + row] = w[1];
    }
    __syncthreads();

    const int kb_ = (lane >> 4) * 16;
    v16bf afrag[2];
#pragma unroll
    for (int mt = 0; mt < 2; ++mt) {
      int m = wave * 32 + mt * 16 + (lane & 15);
      afrag[mt] = frag_cat(*(const v8bf*)&As[m * 32 + kb_],
                           *(const v8bf*)&As[m * 32 + kb_ + 8]);
    }
#pragma unroll
    for (int nt = 0; nt < 4; ++nt) {
      int nl = nt * 16 + (lane & 15);
      v16bf bfrag = *(const v16bf*)&BsT[nl * 32 + (lane >> 4) * 16];
#pragma unroll
      for (int mt = 0; mt < 2; ++mt)
        acc[mt][nt] = wmma_bf16(afrag[mt], bfrag, acc[mt][nt]);
    }
  }

#pragma unroll
  for (int mt = 0; mt < 2; ++mt)
#pragma unroll
    for (int nt = 0; nt < 4; ++nt)
#pragma unroll
      for (int r = 0; r < 8; ++r) {
        int m   = m0 + wave * 32 + mt * 16 + r + 8 * (lane >> 4);
        int col = n0 + nt * 16 + (lane & 15);
        out[(size_t)m * DMODEL + col] = acc[mt][nt][r] + bias[col];
      }
}

// ---------------------------------------------------------------------------
extern "C" void kernel_launch(void* const* d_in, const int* in_sizes, int n_in,
                              void* d_out, int out_size, void* d_ws, size_t ws_size,
                              hipStream_t stream) {
  (void)in_sizes; (void)n_in; (void)out_size; (void)ws_size;

  const float* x       = (const float*)d_in[0];
  /* d_in[1] = pe, unused by the reference forward */
  const float* W_qkv   = (const float*)d_in[2];
  const float* q_scale = (const float*)d_in[3];
  const float* k_scale = (const float*)d_in[4];
  const float* W_proj  = (const float*)d_in[5];
  const float* b_proj  = (const float*)d_in[6];
  float*       out     = (float*)d_out;

  char* ws = (char*)d_ws;
  const size_t MB = (size_t)1 << 20;
  __bf16* xb     = (__bf16*)(ws + 0 * MB);    //  8 MB: x as bf16 [4096,1024]
  __bf16* wqkvb  = (__bf16*)(ws + 8 * MB);    //  6 MB: W_qkv bf16 [1024,3072]
  __bf16* wprojb = (__bf16*)(ws + 14 * MB);   //  2 MB: W_proj bf16 [1024,1024]
  __bf16* qb     = (__bf16*)(ws + 16 * MB);   //  8 MB: q bf16 [B,H,L,DH]
  __bf16* kb     = (__bf16*)(ws + 24 * MB);   //  8 MB: k bf16 [B,H,L,DH]
  __bf16* vb     = (__bf16*)(ws + 32 * MB);   //  8 MB: v bf16 [B,H,L,DH]
  __bf16* attnb  = (__bf16*)(ws + 40 * MB);   //  8 MB: attn out bf16 [4096,1024]

  cvt_f32_bf16<<<dim3(1024), dim3(256), 0, stream>>>(x, xb, (4096 * 1024) / 4);
  cvt_f32_bf16<<<dim3(1024), dim3(256), 0, stream>>>(W_qkv, wqkvb, (1024 * 3072) / 4);
  cvt_f32_bf16<<<dim3(1024), dim3(256), 0, stream>>>(W_proj, wprojb, (1024 * 1024) / 4);

  attn_qkv_gemm<<<dim3(48, 32), dim3(128), 0, stream>>>(xb, wqkvb, q_scale, k_scale,
                                                        qb, kb, vb);
  attn_flash<<<dim3(32, 32), dim3(128), 0, stream>>>(qb, kb, vb, attnb);
  attn_proj_gemm<<<dim3(16, 32), dim3(128), 0, stream>>>(attnb, wprojb, b_proj, out);
}
